// DiscoveryEngineModel_29111288332337
// MI455X (gfx1250) — matrix operand order repaired and checked by reading, not compiled, not run on hardware
//
#include <hip/hip_runtime.h>

// CDNA5 / gfx1250: wave32, WMMA 16x16x32 bf16 -> f32 accum.
typedef __attribute__((ext_vector_type(16))) __bf16 v16bf;
typedef __attribute__((ext_vector_type(8)))  float  v8f;

#define WAVES 4

__device__ __forceinline__ unsigned short f2bf(float f) {
    unsigned u = __float_as_uint(f);
    u += 0x7FFFu + ((u >> 16) & 1u);   // round-to-nearest-even
    return (unsigned short)(u >> 16);
}

__device__ __forceinline__ float softplus_f(float v) {
    return v > 20.f ? v : log1pf(__expf(v));
}

__device__ __forceinline__ v8f wmma_bf16(v16bf a, v16bf b, v8f c) {
    return __builtin_amdgcn_wmma_f32_16x16x32_bf16(
        /*neg_a=*/false, a, /*neg_b=*/false, b,
        /*c_mod=*/(short)0, c, /*reuse_a=*/false, /*reuse_b=*/false);
}

// A fragment (16x32 bf16) from an LDS row-major [16][rowStride] tile.
// 16-bit A layout: lane m=lane&15; halves 0..7 -> K=k0..k0+7, halves 8..15 -> K=k0+16..k0+23,
// where k0 = kt*32 + (lane>=16 ? 8 : 0). Both runs are 16B-contiguous -> two ds_load_b128.
__device__ __forceinline__ v16bf load_a_frag(const unsigned short* tile, int rowStride,
                                             int kt, int lane) {
    int m  = lane & 15;
    int k0 = kt * 32 + ((lane & 16) ? 8 : 0);
    const unsigned short* p = tile + m * rowStride + k0;
    union { uint4 u[2]; v16bf v; } r;
    r.u[0] = *(const uint4*)(p);
    r.u[1] = *(const uint4*)(p + 16);
    return r.v;
}

// B fragment (32x16 bf16) from the fragment-major packed weight buffer:
// buf[((kt*NT+nt)*32 + lane)*16 + h]  -> two global_load_b128 per lane.
__device__ __forceinline__ v16bf load_b_frag(const unsigned short* __restrict__ buf,
                                             int kt, int nt, int NT, int lane) {
    union { uint4 u[2]; v16bf v; } r;
    const uint4* p = (const uint4*)(buf + ((size_t)((kt * NT + nt) * 32 + lane) << 4));
    r.u[0] = p[0];
    r.u[1] = p[1];
    return r.v;
}

// ---------------- weight packing: f32 row-major [realK][ncols] -> bf16 B-fragments ------------
__global__ void pack_w_bf16(const float* __restrict__ W, int realK, int ncols,
                            int KT, int NT, unsigned short* __restrict__ out) {
    int idx = blockIdx.x * blockDim.x + threadIdx.x;
    int total = KT * NT * 512;
    if (idx >= total) return;
    int h    = idx & 15;
    int lane = (idx >> 4) & 31;
    int f    = idx >> 9;
    int nt = f % NT, kt = f / NT;
    int k = kt * 32 + ((lane >> 4) << 3) + (h & 7) + ((h >> 3) << 4);   // lanes 0-15: K 0..7,16..23; lanes 16-31: K 8..15,24..31
    int n = nt * 16 + (lane & 15);
    float v = (k < realK && n < ncols) ? W[(size_t)k * ncols + n] : 0.f;
    out[idx] = f2bf(v);
}

__global__ void zero_f32(float* __restrict__ p, int n) {
    int i = blockIdx.x * blockDim.x + threadIdx.x;
    if (i < n) p[i] = 0.f;
}

// ---------------- edge kernel: phi_e + phi_v + scatter-add ----------------
__global__ __launch_bounds__(128) void egnn_edge_kernel(
    const float* __restrict__ x, const float* __restrict__ pos, const float* __restrict__ vel,
    const int* __restrict__ edge_index,
    const unsigned short* __restrict__ We1p, const float* __restrict__ be1,
    const float* __restrict__ ge1, const float* __restrict__ bte1,
    const unsigned short* __restrict__ We2p, const float* __restrict__ be2,
    const unsigned short* __restrict__ We3p, const float* __restrict__ be3,
    const unsigned short* __restrict__ Wv1p, const float* __restrict__ bv1,
    const float* __restrict__ gv1, const float* __restrict__ btv1,
    const float* __restrict__ Wv2, const float* __restrict__ bv2,
    float* __restrict__ mh_sum, float* __restrict__ mv_sum, float* __restrict__ cnt,
    int E)
{
    __shared__ __attribute__((aligned(16))) unsigned short s_tmp[WAVES][16][160]; // edge feats bf16
    __shared__ __attribute__((aligned(16))) unsigned short s_h[WAVES][16][128];   // hidden bf16
    __shared__ float s_geo[WAVES][16][2];
    __shared__ float s_vw[WAVES][16];
    __shared__ int   s_dst[WAVES][16];

    const int wave = threadIdx.x >> 5;
    const int lane = threadIdx.x & 31;
    const int tile = blockIdx.x * WAVES + wave;
    const bool active = (tile * 16) < E;
    const int ebase = tile * 16;
    const int* __restrict__ srcI = edge_index;
    const int* __restrict__ dstI = edge_index + E;

    const int colb   = lane & 15;
    const int rowoff = (lane >> 4) << 3;

    // ---- fill feature tile: [x_dst(64) | x_src(64) | d2,dv,r2,r6,r12 | pad->160] ----
    if (active) {
        if (lane < 16) {
            int e = ebase + lane;
            int js = srcI[e], jd = dstI[e];
            s_dst[wave][lane] = jd;
            float rpx = pos[2*js]   - pos[2*jd];
            float rpy = pos[2*js+1] - pos[2*jd+1];
            float rvx = vel[2*js]   - vel[2*jd];
            float rvy = vel[2*js+1] - vel[2*jd+1];
            float d2  = rpx*rpx + rpy*rpy;
            float dv  = rvx*rpx + rvy*rpy;
            float r2  = fminf(1.f / (d2 + 0.05f), 20.f);
            float r6  = fminf(r2*r2*r2, 400.f);
            float r12 = fminf(r6*r6, 160000.f);
            s_geo[wave][lane][0] = rpx;
            s_geo[wave][lane][1] = rpy;
            unsigned short* row = &s_tmp[wave][lane][0];
            row[128] = f2bf(d2); row[129] = f2bf(dv); row[130] = f2bf(r2);
            row[131] = f2bf(r6); row[132] = f2bf(r12);
            #pragma unroll
            for (int k = 133; k < 160; ++k) row[k] = 0;
        }
        for (int i = lane; i < 16 * 32; i += 32) {     // 16 edges x 32 float4 chunks
            int m = i >> 5, c = i & 31;
            int node = (c < 16) ? dstI[ebase + m] : srcI[ebase + m];
            int cc = c & 15;
            float4 xv = ((const float4*)(x + (size_t)node * 64))[cc];
            unsigned short* row = &s_tmp[wave][m][(c < 16 ? 0 : 64) + cc * 4];
            uint2 pk;
            pk.x = (unsigned)f2bf(xv.x) | ((unsigned)f2bf(xv.y) << 16);
            pk.y = (unsigned)f2bf(xv.z) | ((unsigned)f2bf(xv.w) << 16);
            *(uint2*)row = pk;
        }
    }
    __syncthreads();  // S1

    v8f acc_e[8], acc_v[8];
    if (active) {
        // ---- GEMM1 (phi_e layer1) and phi_v layer1, sharing A fragments ----
        #pragma unroll
        for (int nt = 0; nt < 8; ++nt) { acc_e[nt] = (v8f)0.f; acc_v[nt] = (v8f)0.f; }
        #pragma unroll
        for (int kt = 0; kt < 5; ++kt) {
            v16bf a = load_a_frag(&s_tmp[wave][0][0], 160, kt, lane);
            #pragma unroll
            for (int nt = 0; nt < 8; ++nt) {
                acc_e[nt] = wmma_bf16(a, load_b_frag(We1p, kt, nt, 8, lane), acc_e[nt]);
                acc_v[nt] = wmma_bf16(a, load_b_frag(Wv1p, kt, nt, 8, lane), acc_v[nt]);
            }
        }
        #pragma unroll
        for (int nt = 0; nt < 8; ++nt) {
            float b1 = be1[nt * 16 + colb];
            float bv = bv1[nt * 16 + colb];
            #pragma unroll
            for (int r = 0; r < 8; ++r) { acc_e[nt][r] += b1; acc_v[nt][r] += bv; }
        }
        // ---- LayerNorm + softplus on acc_e (rows split across lane halves), write h1 ----
        float gc[8], bc[8];
        #pragma unroll
        for (int nt = 0; nt < 8; ++nt) { gc[nt] = ge1[nt*16 + colb]; bc[nt] = bte1[nt*16 + colb]; }
        #pragma unroll
        for (int r = 0; r < 8; ++r) {
            float s = 0.f, q = 0.f;
            #pragma unroll
            for (int nt = 0; nt < 8; ++nt) { float v = acc_e[nt][r]; s += v; q += v * v; }
            #pragma unroll
            for (int o = 1; o < 16; o <<= 1) { s += __shfl_xor(s, o, 32); q += __shfl_xor(q, o, 32); }
            float mu = s * (1.f / 128.f);
            float rs = rsqrtf(q * (1.f / 128.f) - mu * mu + 1e-5f);
            #pragma unroll
            for (int nt = 0; nt < 8; ++nt) {
                float v = (acc_e[nt][r] - mu) * rs * gc[nt] + bc[nt];
                s_h[wave][rowoff + r][nt * 16 + colb] = f2bf(softplus_f(v));
            }
        }
    }
    __syncthreads();  // S2

    if (active) {
        // ---- phi_v tail: LN + softplus + dot(Wv2) via half-wave reduce ----
        {
            float gc[8], bc[8], wv[8];
            #pragma unroll
            for (int nt = 0; nt < 8; ++nt) {
                gc[nt] = gv1[nt*16 + colb]; bc[nt] = btv1[nt*16 + colb]; wv[nt] = Wv2[nt*16 + colb];
            }
            #pragma unroll
            for (int r = 0; r < 8; ++r) {
                float s = 0.f, q = 0.f;
                #pragma unroll
                for (int nt = 0; nt < 8; ++nt) { float v = acc_v[nt][r]; s += v; q += v * v; }
                #pragma unroll
                for (int o = 1; o < 16; o <<= 1) { s += __shfl_xor(s, o, 32); q += __shfl_xor(q, o, 32); }
                float mu = s * (1.f / 128.f);
                float rs = rsqrtf(q * (1.f / 128.f) - mu * mu + 1e-5f);
                float t = 0.f;
                #pragma unroll
                for (int nt = 0; nt < 8; ++nt) {
                    float v = (acc_v[nt][r] - mu) * rs * gc[nt] + bc[nt];
                    t += softplus_f(v) * wv[nt];
                }
                #pragma unroll
                for (int o = 1; o < 16; o <<= 1) t += __shfl_xor(t, o, 32);
                if (colb == 0) s_vw[wave][rowoff + r] = t + bv2[0];
            }
        }
        // ---- GEMM2 (phi_e layer2), reuse acc_e ----
        #pragma unroll
        for (int nt = 0; nt < 8; ++nt) acc_e[nt] = (v8f)0.f;
        #pragma unroll
        for (int kt = 0; kt < 4; ++kt) {
            v16bf a = load_a_frag(&s_h[wave][0][0], 128, kt, lane);
            #pragma unroll
            for (int nt = 0; nt < 8; ++nt)
                acc_e[nt] = wmma_bf16(a, load_b_frag(We2p, kt, nt, 8, lane), acc_e[nt]);
        }
    }
    __syncthreads();  // S3

    if (active) {  // overwrite h tile with softplus(h1@We2 + be2)
        #pragma unroll
        for (int nt = 0; nt < 8; ++nt) {
            float b2 = be2[nt * 16 + colb];
            #pragma unroll
            for (int r = 0; r < 8; ++r)
                s_h[wave][rowoff + r][nt * 16 + colb] = f2bf(softplus_f(acc_e[nt][r] + b2));
        }
    }
    __syncthreads();  // S4

    if (active) {
        // ---- GEMM3 (phi_e layer3), reuse acc_v; scatter-add m_h_e ----
        #pragma unroll
        for (int nt = 0; nt < 8; ++nt) acc_v[nt] = (v8f)0.f;
        #pragma unroll
        for (int kt = 0; kt < 4; ++kt) {
            v16bf a = load_a_frag(&s_h[wave][0][0], 128, kt, lane);
            #pragma unroll
            for (int nt = 0; nt < 8; ++nt)
                acc_v[nt] = wmma_bf16(a, load_b_frag(We3p, kt, nt, 8, lane), acc_v[nt]);
        }
        #pragma unroll
        for (int nt = 0; nt < 8; ++nt) {
            float b3 = be3[nt * 16 + colb];
            #pragma unroll
            for (int r = 0; r < 8; ++r) {
                int jd = s_dst[wave][rowoff + r];
                atomicAdd(&mh_sum[(size_t)jd * 128 + nt * 16 + colb], acc_v[nt][r] + b3);
            }
        }
        if (lane < 16) {  // equivariant message + counts
            float vw = s_vw[wave][lane];
            int   jd = s_dst[wave][lane];
            atomicAdd(&mv_sum[2 * jd],     vw * s_geo[wave][lane][0]);
            atomicAdd(&mv_sum[2 * jd + 1], vw * s_geo[wave][lane][1]);
            atomicAdd(&cnt[jd], 1.f);
        }
    }
}

// ---------------- node kernel: phi_h + residual ----------------
__global__ __launch_bounds__(128) void egnn_node_kernel(
    const float* __restrict__ x,
    const float* __restrict__ mh_sum, const float* __restrict__ mv_sum, const float* __restrict__ cnt,
    const unsigned short* __restrict__ Wh1p, const float* __restrict__ bh1,
    const float* __restrict__ gh1, const float* __restrict__ bth1,
    const unsigned short* __restrict__ Wh2p, const float* __restrict__ bh2,
    float* __restrict__ out, int N)
{
    __shared__ __attribute__((aligned(16))) unsigned short s_tmp[WAVES][16][224]; // [x|m_h|norm|pad]
    __shared__ __attribute__((aligned(16))) unsigned short s_h[WAVES][16][128];

    const int wave = threadIdx.x >> 5;
    const int lane = threadIdx.x & 31;
    const int tile = blockIdx.x * WAVES + wave;
    const bool active = (tile * 16) < N;
    const int nbase = tile * 16;
    const int colb   = lane & 15;
    const int rowoff = (lane >> 4) << 3;

    if (active) {
        if (lane < 16) {
            int n = nbase + lane;
            float dn  = fmaxf(cnt[n], 1.f);
            float mvx = mv_sum[2 * n]     / dn + 1e-8f;
            float mvy = mv_sum[2 * n + 1] / dn + 1e-8f;
            float nrm = sqrtf(mvx * mvx + mvy * mvy);
            unsigned short* row = &s_tmp[wave][lane][0];
            row[192] = f2bf(nrm);
            #pragma unroll
            for (int k = 193; k < 224; ++k) row[k] = 0;
        }
        for (int i = lane; i < 16 * 16; i += 32) {     // x: 16 nodes x 16 float4
            int m = i >> 4, cc = i & 15;
            int node = nbase + m;
            float4 xv = ((const float4*)(x + (size_t)node * 64))[cc];
            unsigned short* row = &s_tmp[wave][m][cc * 4];
            uint2 pk;
            pk.x = (unsigned)f2bf(xv.x) | ((unsigned)f2bf(xv.y) << 16);
            pk.y = (unsigned)f2bf(xv.z) | ((unsigned)f2bf(xv.w) << 16);
            *(uint2*)row = pk;
        }
        for (int i = lane; i < 16 * 32; i += 32) {     // m_h mean: 16 nodes x 32 float4
            int m = i >> 5, cc = i & 31;
            int node = nbase + m;
            float inv = 1.f / fmaxf(cnt[node], 1.f);
            float4 hv = ((const float4*)(mh_sum + (size_t)node * 128))[cc];
            unsigned short* row = &s_tmp[wave][m][64 + cc * 4];
            uint2 pk;
            pk.x = (unsigned)f2bf(hv.x * inv) | ((unsigned)f2bf(hv.y * inv) << 16);
            pk.y = (unsigned)f2bf(hv.z * inv) | ((unsigned)f2bf(hv.w * inv) << 16);
            *(uint2*)row = pk;
        }
    }
    __syncthreads();  // S1

    v8f acc[8];
    if (active) {
        #pragma unroll
        for (int nt = 0; nt < 8; ++nt) acc[nt] = (v8f)0.f;
        #pragma unroll
        for (int kt = 0; kt < 7; ++kt) {
            v16bf a = load_a_frag(&s_tmp[wave][0][0], 224, kt, lane);
            #pragma unroll
            for (int nt = 0; nt < 8; ++nt)
                acc[nt] = wmma_bf16(a, load_b_frag(Wh1p, kt, nt, 8, lane), acc[nt]);
        }
        float gc[8], bc[8];
        #pragma unroll
        for (int nt = 0; nt < 8; ++nt) {
            float b1 = bh1[nt * 16 + colb];
            #pragma unroll
            for (int r = 0; r < 8; ++r) acc[nt][r] += b1;
            gc[nt] = gh1[nt * 16 + colb];
            bc[nt] = bth1[nt * 16 + colb];
        }
        #pragma unroll
        for (int r = 0; r < 8; ++r) {
            float s = 0.f, q = 0.f;
            #pragma unroll
            for (int nt = 0; nt < 8; ++nt) { float v = acc[nt][r]; s += v; q += v * v; }
            #pragma unroll
            for (int o = 1; o < 16; o <<= 1) { s += __shfl_xor(s, o, 32); q += __shfl_xor(q, o, 32); }
            float mu = s * (1.f / 128.f);
            float rs = rsqrtf(q * (1.f / 128.f) - mu * mu + 1e-5f);
            #pragma unroll
            for (int nt = 0; nt < 8; ++nt) {
                float v = (acc[nt][r] - mu) * rs * gc[nt] + bc[nt];
                s_h[wave][rowoff + r][nt * 16 + colb] = f2bf(softplus_f(v));
            }
        }
    }
    __syncthreads();  // S2

    if (active) {
        v8f acc2[4];
        #pragma unroll
        for (int nt = 0; nt < 4; ++nt) acc2[nt] = (v8f)0.f;
        #pragma unroll
        for (int kt = 0; kt < 4; ++kt) {
            v16bf a = load_a_frag(&s_h[wave][0][0], 128, kt, lane);
            #pragma unroll
            for (int nt = 0; nt < 4; ++nt)
                acc2[nt] = wmma_bf16(a, load_b_frag(Wh2p, kt, nt, 4, lane), acc2[nt]);
        }
        #pragma unroll
        for (int nt = 0; nt < 4; ++nt) {
            int col = nt * 16 + colb;
            float b2 = bh2[col];
            #pragma unroll
            for (int r = 0; r < 8; ++r) {
                int node = nbase + rowoff + r;
                out[(size_t)node * 64 + col] = x[(size_t)node * 64 + col] + acc2[nt][r] + b2;
            }
        }
    }
}

extern "C" void kernel_launch(void* const* d_in, const int* in_sizes, int n_in,
                              void* d_out, int out_size, void* d_ws, size_t ws_size,
                              hipStream_t stream) {
    const float* x    = (const float*)d_in[0];
    const float* pos  = (const float*)d_in[1];
    const float* vel  = (const float*)d_in[2];
    const float* We1  = (const float*)d_in[3];
    const float* be1  = (const float*)d_in[4];
    const float* ge1  = (const float*)d_in[5];
    const float* bte1 = (const float*)d_in[6];
    const float* We2  = (const float*)d_in[7];
    const float* be2  = (const float*)d_in[8];
    const float* We3  = (const float*)d_in[9];
    const float* be3  = (const float*)d_in[10];
    const float* Wv1  = (const float*)d_in[11];
    const float* bv1  = (const float*)d_in[12];
    const float* gv1  = (const float*)d_in[13];
    const float* btv1 = (const float*)d_in[14];
    const float* Wv2  = (const float*)d_in[15];
    const float* bv2  = (const float*)d_in[16];
    const float* Wh1  = (const float*)d_in[17];
    const float* bh1  = (const float*)d_in[18];
    const float* gh1  = (const float*)d_in[19];
    const float* bth1 = (const float*)d_in[20];
    const float* Wh2  = (const float*)d_in[21];
    const float* bh2  = (const float*)d_in[22];
    const int* edge_index = (const int*)d_in[23];

    const int N = in_sizes[0] / 64;
    const int E = in_sizes[23] / 2;

    // workspace carve-up (256B aligned)
    unsigned char* ws = (unsigned char*)d_ws;
    size_t o = 0;
    auto carve = [&](size_t bytes) -> void* {
        o = (o + 255) & ~(size_t)255;
        void* p = ws + o;
        o += bytes;
        return p;
    };
    unsigned short* We1p = (unsigned short*)carve(5 * 8 * 512 * 2);
    unsigned short* Wv1p = (unsigned short*)carve(5 * 8 * 512 * 2);
    unsigned short* We2p = (unsigned short*)carve(4 * 8 * 512 * 2);
    unsigned short* We3p = (unsigned short*)carve(4 * 8 * 512 * 2);
    unsigned short* Wh1p = (unsigned short*)carve(7 * 8 * 512 * 2);
    unsigned short* Wh2p = (unsigned short*)carve(4 * 4 * 512 * 2);
    const size_t accFloats = (size_t)N * 128 + (size_t)N * 2 + (size_t)N;
    float* mh_sum = (float*)carve(accFloats * 4);
    float* mv_sum = mh_sum + (size_t)N * 128;
    float* cntp   = mv_sum + (size_t)N * 2;

    // pack weights into WMMA-B fragment layout (bf16)
    auto packs = [&](const float* W, int realK, int ncols, int KT, int NT, unsigned short* outp) {
        int total = KT * NT * 512;
        pack_w_bf16<<<(total + 255) / 256, 256, 0, stream>>>(W, realK, ncols, KT, NT, outp);
    };
    packs(We1, 133, 128, 5, 8, We1p);
    packs(Wv1, 133, 128, 5, 8, Wv1p);
    packs(We2, 128, 128, 4, 8, We2p);
    packs(We3, 128, 128, 4, 8, We3p);
    packs(Wh1, 193, 128, 7, 8, Wh1p);
    packs(Wh2, 128,  64, 4, 4, Wh2p);

    zero_f32<<<(int)((accFloats + 255) / 256), 256, 0, stream>>>(mh_sum, (int)accFloats);

    const int tilesE  = (E + 15) / 16;
    const int blocksE = (tilesE + WAVES - 1) / WAVES;
    egnn_edge_kernel<<<blocksE, 128, 0, stream>>>(
        x, pos, vel, edge_index,
        We1p, be1, ge1, bte1,
        We2p, be2, We3p, be3,
        Wv1p, bv1, gv1, btv1, Wv2, bv2,
        mh_sum, mv_sum, cntp, E);

    const int tilesN  = (N + 15) / 16;
    const int blocksN = (tilesN + WAVES - 1) / WAVES;
    egnn_node_kernel<<<blocksN, 128, 0, stream>>>(
        x, mh_sum, mv_sum, cntp,
        Wh1p, bh1, gh1, bth1,
        Wh2p, bh2,
        (float*)d_out, N);
}
